// TriangleMultiplication_515396076357
// MI455X (gfx1250) — matrix operand order repaired and checked by reading, not compiled, not run on hardware
//
#include <hip/hip_runtime.h>

// ---------------------------------------------------------------------------
// TriangleMultiplication (ENDING=True), B=1, N=512, D=128, fp32 in/out.
// Strategy: bf16 WMMA (v_wmma_f32_16x16x32_bf16) for all three GEMM stages,
// fp32 accumulate. a/b stored bf16 channel-major so the triangle GEMM
// (t[i,j,d] = sum_k a[k,i,d]*b[k,j,d]) is 128 per-channel 512x512x512 GEMMs
// with contiguous per-lane fragment loads; both buffers (134MB total) target
// the 192MB L2. k2 uses 32x64 register blocking per wave so each K-chunk does
// 8 WMMAs per 12 b128 loads (B fragments reused across two i-tiles).
// ---------------------------------------------------------------------------

#define NN 512
#define DD 128

typedef __attribute__((ext_vector_type(16))) __bf16          v16bf;
typedef __attribute__((ext_vector_type(8)))  float           v8f;
typedef __attribute__((ext_vector_type(4)))  float           f4;
typedef __attribute__((ext_vector_type(8)))  unsigned short  us8;
typedef __attribute__((ext_vector_type(16))) unsigned short  us16;

__device__ __forceinline__ unsigned short f2bf(float f) {
  unsigned u = __float_as_uint(f);
  u += 0x7FFFu + ((u >> 16) & 1u);      // round-to-nearest-even
  return (unsigned short)(u >> 16);
}

__device__ __forceinline__ float sigmoidf(float x) {
  return 1.0f / (1.0f + __expf(-x));
}

__device__ __forceinline__ float wred32(float v) {
#pragma unroll
  for (int o = 16; o > 0; o >>= 1) v += __shfl_xor(v, o, 32);
  return v;
}

__device__ __forceinline__ v8f wmma_bf16(us16 a, us16 b, v8f c) {
  return __builtin_amdgcn_wmma_f32_16x16x32_bf16(
      false, __builtin_bit_cast(v16bf, a),
      false, __builtin_bit_cast(v16bf, b),
      (short)0, c, false, false);
}

// A fragment: 16(M) x 32(K) bf16 from row-major source (rowStride elems).
// Contiguous-k layout (same as load_b_frag): lanes 0-15 = row m, K 0..15; lanes 16-31 = K 16..31.
__device__ __forceinline__ us16 load_a_frag(const unsigned short* src, int rowStride) {
  const int l  = threadIdx.x & 31;
  const int m  = l & 15;
  const int ko = (l < 16) ? 0 : 16;
  const unsigned short* p = src + m * rowStride + ko;
  us8 lo = *(const us8*)(p);
  us8 hi = *(const us8*)(p + 8);
  us16 r;
#pragma unroll
  for (int i = 0; i < 8; ++i) { r[i] = lo[i]; r[8 + i] = hi[i]; }
  return r;
}

// B fragment: 32(K) x 16(N) bf16 where each column has K contiguous in memory
// (colStride elems between columns). lanes 0-15: col n, K 0..15; lanes 16-31: K 16..31.
__device__ __forceinline__ us16 load_b_frag(const unsigned short* colBase, int colStride) {
  const int l  = threadIdx.x & 31;
  const int n  = l & 15;
  const int ko = (l < 16) ? 0 : 16;
  return *(const us16*)(colBase + n * colStride + ko);
}

// ---------------------------------------------------------------------------
// k0: convert & concat weights to bf16: rows [0,256)=g_in, [256,512)=p_in,
//     [512,640)=g_out, [640,768)=p_out. 768*128 elems.
// ---------------------------------------------------------------------------
__global__ __launch_bounds__(256) void k0_weights(
    const float* __restrict__ g_in, const float* __restrict__ p_in,
    const float* __restrict__ g_out, const float* __restrict__ p_out,
    unsigned short* __restrict__ Wcat) {
  int idx = blockIdx.x * 256 + threadIdx.x;
  if (idx >= 768 * DD) return;
  int row = idx / DD;
  float v;
  if      (row < 256) v = g_in[idx];
  else if (row < 512) v = p_in[idx - 256 * DD];
  else if (row < 640) v = g_out[idx - 512 * DD];
  else                v = p_out[idx - 640 * DD];
  Wcat[idx] = f2bf(v);
}

// ---------------------------------------------------------------------------
// k1: per block: rows (p0+m, q) for m=0..15 (tile runs along the k-axis of the
// triangle einsum). LN_in -> xn(bf16, LDS) -> 16x640 WMMA GEMM -> fused
// sigmoid/mul/mask epilogue scattering into channel-major a/b + g_out store.
//   Abuf[d][q][p] = a[p,q,d]  (= A_d row-major: (i=q, k=p), row stride 512)
//   Bbuf[d][q][p] = b[p,q,d]  (= B_d col-major: col j=q has K contiguous)
// ---------------------------------------------------------------------------
__global__ __launch_bounds__(256) void k1_proj(
    const float* __restrict__ x, const float* __restrict__ mask,
    const float* __restrict__ niw, const float* __restrict__ nib,
    const unsigned short* __restrict__ Wcat,
    unsigned short* __restrict__ Abuf, unsigned short* __restrict__ Bbuf,
    float* __restrict__ Gout) {
  const int q  = blockIdx.x;
  const int p0 = blockIdx.y * 16;
  __shared__ unsigned short xn[16][DD];   // 4KB bf16 normalized tile
  __shared__ float outT[16][640];         // 40KB GEMM results
  __shared__ float msk[16];
  const int tid = threadIdx.x, lane = tid & 31, wv = tid >> 5;

  if (tid < 16) msk[tid] = mask[(size_t)(p0 + tid) * NN + q];

  // LayerNorm: wave wv handles rows wv and wv+8, one row per wave (4 elems/lane)
#pragma unroll
  for (int rr = 0; rr < 2; ++rr) {
    const int m = wv + rr * 8;
    const float* xr = x + ((size_t)(p0 + m) * NN + q) * DD;
    f4 v = *(const f4*)(xr + lane * 4);
    float s = wred32(v[0] + v[1] + v[2] + v[3]);
    float mu = s * (1.0f / 128.0f);
    float d0 = v[0] - mu, d1 = v[1] - mu, d2 = v[2] - mu, d3 = v[3] - mu;
    float s2 = wred32(d0 * d0 + d1 * d1 + d2 * d2 + d3 * d3);
    float inv = rsqrtf(s2 * (1.0f / 128.0f) + 1e-5f);
#pragma unroll
    for (int t = 0; t < 4; ++t) {
      int c = lane * 4 + t;
      float xv = (v[t] - mu) * inv * niw[c] + nib[c];
      xn[m][c] = f2bf(xv);
    }
  }
  __syncthreads();

  // GEMM: out[16][640] = xn[16][128] @ W^T ; wave wv owns col tiles wv*5..wv*5+4
  v8f acc[5];
#pragma unroll
  for (int tt = 0; tt < 5; ++tt) acc[tt] = (v8f){0, 0, 0, 0, 0, 0, 0, 0};
#pragma unroll
  for (int kc = 0; kc < 4; ++kc) {
    us16 af = load_a_frag(&xn[0][0] + kc * 32, DD);
#pragma unroll
    for (int tt = 0; tt < 5; ++tt) {
      const int e0 = (wv * 5 + tt) * 16;
      us16 bf_ = load_b_frag(Wcat + (size_t)e0 * DD + kc * 32, DD);
      acc[tt] = wmma_bf16(af, bf_, acc[tt]);
    }
  }
  // C layout: vgpr r -> (M = r + (lane<16?0:8), N = lane&15)
  {
    const int nn = lane & 15, hi = (lane < 16) ? 0 : 8;
#pragma unroll
    for (int tt = 0; tt < 5; ++tt) {
      const int e0 = (wv * 5 + tt) * 16;
#pragma unroll
      for (int r = 0; r < 8; ++r) outT[r + hi][e0 + nn] = acc[tt][r];
    }
  }
  __syncthreads();

  // Epilogue A/B: thread t -> (array = t>>7, channel d = t&127); 16 rows -> one 32B store
  {
    const int half = tid >> 7;     // 0 -> a, 1 -> b
    const int d    = tid & 127;
    us16 seg;
#pragma unroll
    for (int m = 0; m < 16; ++m) {
      float g = outT[m][half * DD + d];          // gate logits (cols 0..255)
      float p = outT[m][256 + half * DD + d];    // linear      (cols 256..511)
      seg[m] = f2bf(sigmoidf(g) * p * msk[m]);
    }
    unsigned short* dst = (half ? Bbuf : Abuf) +
                          (size_t)d * NN * NN + (size_t)q * NN + p0;
    *(us16*)dst = seg;
  }
  // Epilogue g_out (stored post-sigmoid, fp32 row-major [pos][e])
  {
    const int m = tid >> 4, e0 = (tid & 15) * 8;
    float* dst = Gout + ((size_t)(p0 + m) * NN + q) * DD + e0;
    f4 lo, hi;
#pragma unroll
    for (int u = 0; u < 4; ++u) lo[u] = sigmoidf(outT[m][512 + e0 + u]);
#pragma unroll
    for (int u = 0; u < 4; ++u) hi[u] = sigmoidf(outT[m][512 + e0 + 4 + u]);
    *(f4*)dst = lo;
    *(f4*)(dst + 4) = hi;
  }
}

// ---------------------------------------------------------------------------
// k2: triangle GEMM. Block = 4 waves = 4 consecutive channels; each wave:
// one d, 32(i) x 64(j) output block (2 i-tiles x 4 j-tiles, 64 acc VGPRs),
// K=512 in 32-steps. B fragments are reused across both i-tiles: 8 WMMAs per
// 12 b128 loads per K-chunk. T stored [d][i][j] fp32.
// Grid: x = j-block(8), y = i-block(16), z = d-block(32) -> d slowest so one
// d-block's 128 workgroups co-reside touching only ~4MB of a/b (L2-hot).
// ---------------------------------------------------------------------------
__global__ __launch_bounds__(128) void k2_tri(
    const unsigned short* __restrict__ Abuf,
    const unsigned short* __restrict__ Bbuf,
    float* __restrict__ T) {
  const int lane = threadIdx.x & 31, wv = threadIdx.x >> 5;
  const int d  = blockIdx.z * 4 + wv;
  const int i0 = blockIdx.y * 32;
  const int j0 = blockIdx.x * 64;
  const unsigned short* Ad = Abuf + (size_t)d * NN * NN;
  const unsigned short* Bd = Bbuf + (size_t)d * NN * NN;

  v8f acc[2][4];
#pragma unroll
  for (int ii = 0; ii < 2; ++ii)
#pragma unroll
    for (int tt = 0; tt < 4; ++tt) acc[ii][tt] = (v8f){0, 0, 0, 0, 0, 0, 0, 0};

  for (int k0 = 0; k0 < NN; k0 += 32) {
    us16 af0 = load_a_frag(Ad + (size_t)i0 * NN + k0, NN);
    us16 af1 = load_a_frag(Ad + (size_t)(i0 + 16) * NN + k0, NN);
#pragma unroll
    for (int tt = 0; tt < 4; ++tt) {
      us16 bf_ = load_b_frag(Bd + (size_t)(j0 + tt * 16) * NN + k0, NN);
      acc[0][tt] = wmma_bf16(af0, bf_, acc[0][tt]);
      acc[1][tt] = wmma_bf16(af1, bf_, acc[1][tt]);
    }
  }

  float* Td = T + (size_t)d * NN * NN;
  const int nn = lane & 15, hi = (lane < 16) ? 0 : 8;
#pragma unroll
  for (int ii = 0; ii < 2; ++ii) {
#pragma unroll
    for (int tt = 0; tt < 4; ++tt) {
#pragma unroll
      for (int r = 0; r < 8; ++r)
        Td[(size_t)(i0 + ii * 16 + r + hi) * NN + (j0 + tt * 16 + nn)] =
            acc[ii][tt][r];
    }
  }
}

// ---------------------------------------------------------------------------
// k3: per block: 16x16 (i,j) positions. Single-pass LN over d (one thread per
// position; coalesced across d since T is [d][i][j]) -> bf16 tile in dynamic
// LDS -> 256x128 WMMA projection with p_out -> fused * g_out -> fp32 out.
// ---------------------------------------------------------------------------
__global__ __launch_bounds__(256) void k3_out(
    const float* __restrict__ T, const float* __restrict__ Gout,
    const unsigned short* __restrict__ Wcat,
    const float* __restrict__ now, const float* __restrict__ nob,
    float* __restrict__ Out) {
  const int i0 = blockIdx.y * 16, j0 = blockIdx.x * 16;
  extern __shared__ unsigned short tb[];  // [256][136] bf16 (row stride 136 keeps 16B align)
  const int TBS = 136;
  const int tid = threadIdx.x, lane = tid & 31, wv = tid >> 5;

  // LN over d (mean / E[x^2]); thread owns one position, re-reads via L2.
  {
    const int m = tid >> 4, n = tid & 15;
    const float* src = T + (size_t)(i0 + m) * NN + (j0 + n);
    float s = 0.f, s2 = 0.f;
    for (int d = 0; d < DD; ++d) {
      float v = src[(size_t)d * NN * NN];
      s += v; s2 += v * v;
    }
    float mu  = s * (1.0f / 128.0f);
    float var = s2 * (1.0f / 128.0f) - mu * mu;
    float inv = rsqrtf(var + 1e-5f);
    unsigned short* row = tb + (m * 16 + n) * TBS;
    for (int d = 0; d < DD; ++d) {
      float v = src[(size_t)d * NN * NN];
      row[d] = f2bf((v - mu) * inv * now[d] + nob[d]);
    }
  }
  __syncthreads();

  // GEMM 256x128 (K=128): wave wv owns col tile e0 = wv*16; loop 16 row tiles.
  const unsigned short* Wp = Wcat + (size_t)640 * DD;  // p_out rows
  const int nn = lane & 15, hi = (lane < 16) ? 0 : 8;
  for (int rt = 0; rt < 16; ++rt) {
    v8f acc = (v8f){0, 0, 0, 0, 0, 0, 0, 0};
    const unsigned short* atile = tb + (rt * 16) * TBS;
#pragma unroll
    for (int kc = 0; kc < 4; ++kc) {
      us16 af  = load_a_frag(atile + kc * 32, TBS);
      us16 bf_ = load_b_frag(Wp + (size_t)(wv * 16) * DD + kc * 32, DD);
      acc = wmma_bf16(af, bf_, acc);
    }
#pragma unroll
    for (int r = 0; r < 8; ++r) {
      const int pos = rt * 16 + r + hi;
      const int ii = i0 + (pos >> 4), jj = j0 + (pos & 15);
      const size_t off = ((size_t)ii * NN + jj) * DD + wv * 16 + nn;
      Out[off] = acc[r] * Gout[off];
    }
  }
}

// ---------------------------------------------------------------------------
// Host launcher. Workspace layout (bytes):
//   [0,196608)                 Wcat   (768x128 bf16)
//   +67108864                  Abuf   (128*512*512 bf16)
//   +67108864                  Bbuf
//   +134217728                 Gout   (fp32, sigmoided)
//   +134217728                 T      (fp32, [d][i][j])
// total ~403 MB.
// ---------------------------------------------------------------------------
extern "C" void kernel_launch(void* const* d_in, const int* in_sizes, int n_in,
                              void* d_out, int out_size, void* d_ws, size_t ws_size,
                              hipStream_t stream) {
  (void)in_sizes; (void)n_in; (void)out_size; (void)ws_size;
  const float* x    = (const float*)d_in[0];
  const float* mask = (const float*)d_in[1];
  const float* niw  = (const float*)d_in[2];
  const float* nib  = (const float*)d_in[3];
  const float* giw  = (const float*)d_in[4];
  const float* piw  = (const float*)d_in[5];
  const float* now  = (const float*)d_in[6];
  const float* nob  = (const float*)d_in[7];
  const float* gow  = (const float*)d_in[8];
  const float* pow_w = (const float*)d_in[9];
  float* out = (float*)d_out;

  unsigned char* ws = (unsigned char*)d_ws;
  size_t off = 0;
  unsigned short* Wcat = (unsigned short*)(ws + off); off += 196608;
  unsigned short* Abuf = (unsigned short*)(ws + off); off += 67108864;
  unsigned short* Bbuf = (unsigned short*)(ws + off); off += 67108864;
  float*          Gout = (float*)(ws + off);          off += 134217728;
  float*          T    = (float*)(ws + off);

  k0_weights<<<384, 256, 0, stream>>>(giw, piw, gow, pow_w, Wcat);
  k1_proj<<<dim3(512, 32), 256, 0, stream>>>(x, mask, niw, nib, Wcat, Abuf, Bbuf, Gout);
  k2_tri<<<dim3(8, 16, 32), 128, 0, stream>>>(Abuf, Bbuf, T);
  k3_out<<<dim3(32, 32), 256, 256 * 136 * 2, stream>>>(T, Gout, Wcat, now, nob, out);
}
